// HierarchicalPerceiverDecoder_70007966925289
// MI455X (gfx1250) — matrix operand order repaired and checked by reading, not compiled, not run on hardware
//
#include <hip/hip_runtime.h>
#include <hip/hip_bf16.h>
#include <math.h>

// ---------------- model constants (all static) ----------------
#define HD       1280
#define ODIM     768
#define NLAT     256
#define NHEADS   8
#define DHEAD    64
#define DINNER   512
#define NBATCH   2
#define NPTS0    16384
#define OMEGA0f  30.0f
#define ATTNSCALE 0.125f      // 1/sqrt(64)
#define LNEPS    1e-5f

// levels: 16384 -> (cpd=20) 8000 -> (cpd=15) 3375 -> (cpd=11) 1331
static const int NLv[4]  = {16384, 8000, 3375, 1331};
static const int CPDv[3] = {20, 15, 11};

// ---------------- types ----------------
typedef __attribute__((ext_vector_type(16))) __bf16 bf16x16;
typedef __attribute__((ext_vector_type(8)))  __bf16 bf16x8;
typedef __attribute__((ext_vector_type(2)))  __bf16 bf16x2;
typedef __attribute__((ext_vector_type(8)))  float  f32x8;

union FragBF { bf16x16 v; bf16x8 h[2]; unsigned w[8]; unsigned short u[16]; };

__device__ __forceinline__ unsigned short f2bf(float x) {
  unsigned u = __float_as_uint(x);
  return (unsigned short)((u + 0x7fffu + ((u >> 16) & 1u)) >> 16);   // RNE
}
// packed f32x2 -> bf16x2 (hardware v_cvt_pk_bf16_f32 when available)
__device__ __forceinline__ unsigned pack2bf(float x, float y) {
#if __has_builtin(__builtin_amdgcn_cvt_pk_bf16_f32)
  bf16x2 p = __builtin_amdgcn_cvt_pk_bf16_f32(x, y);
  return __builtin_bit_cast(unsigned, p);
#else
  return (unsigned)f2bf(x) | ((unsigned)f2bf(y) << 16);
#endif
}
__device__ __forceinline__ float redmax16(float v) {
#pragma unroll
  for (int m = 1; m < 16; m <<= 1) v = fmaxf(v, __shfl_xor(v, m, 32));
  return v;
}
__device__ __forceinline__ float redsum16(float v) {
#pragma unroll
  for (int m = 1; m < 16; m <<= 1) v += __shfl_xor(v, m, 32);
  return v;
}
__device__ __forceinline__ float redsum32(float v) {
#pragma unroll
  for (int m = 1; m < 32; m <<= 1) v += __shfl_xor(v, m, 32);
  return v;
}

// ---------------- tiled bf16-WMMA GEMM: C = A(MxK) * B(KxN) [+bias][+res] ----
// f32 in memory, converted to bf16 while staging into swizzled, double-buffered
// LDS tiles so next-tile global loads overlap the current tile's WMMAs.
#define BM 128
#define BN 128
#define BK 32

__device__ __forceinline__ void stage_tiles(
    const float* __restrict__ A, const float* __restrict__ B,
    int M, int N, int K, int bm, int bn, int k0, int tid,
    unsigned short* __restrict__ la, unsigned short* __restrict__ lb)
{
  // A tile (BM x BK): float2 loads, packed cvt, xor-swizzled per 8-elem chunk
#pragma unroll
  for (int it = 0; it < 8; ++it) {
    int idx = it * 256 + tid;        // pair index 0..2047
    int row = idx >> 4;
    int col = (idx & 15) << 1;
    int gr = bm + row;
    int cr = gr < M ? gr : M - 1;    // clamp (no exec-mask diamond)
    float2 vv = *(const float2*)(A + (size_t)cr * K + k0 + col);
    if (gr >= M) { vv.x = 0.f; vv.y = 0.f; }
    *(unsigned*)(&la[row * BK + (col ^ ((row & 3) << 3))]) = pack2bf(vv.x, vv.y);
  }
  // B tile transposed (BN x BK)
#pragma unroll
  for (int it = 0; it < 16; ++it) {
    int idx = it * 256 + tid;        // 0..4095
    int n  = idx & 127;
    int k  = idx >> 7;
    int gn = bn + n;
    int cn = gn < N ? gn : N - 1;
    float v = B[(size_t)(k0 + k) * N + cn];
    if (gn >= N) v = 0.f;
    lb[n * BK + (k ^ ((n & 3) << 3))] = f2bf(v);
  }
}

__global__ __launch_bounds__(256)
void k_gemm(const float* __restrict__ A, const float* __restrict__ B,
            const float* __restrict__ bias, const float* __restrict__ res,
            float* __restrict__ C, int M, int N, int K, int flags)
{
  __shared__ unsigned short ldsA[2][BM * BK];
  __shared__ unsigned short ldsB[2][BN * BK];
  const int tid  = threadIdx.x;
  const int wave = tid >> 5;
  const int lane = tid & 31;
  const int r    = lane & 15;
  const int hi   = lane >> 4;
  const int kb   = hi << 3;      // A-frag K base
  const int hi16 = hi << 4;      // unused (B-frag K base is kb too)
  const int bm = blockIdx.x * BM;
  const int bn = blockIdx.y * BN;
  const int wm = (wave & 1) * 64;   // 2 waves along M (64 rows each)
  const int wn = (wave >> 1) * 32;  // 4 waves along N (32 cols each)

  f32x8 acc[4][2];
#pragma unroll
  for (int mi = 0; mi < 4; ++mi)
#pragma unroll
    for (int ni = 0; ni < 2; ++ni)
#pragma unroll
      for (int j = 0; j < 8; ++j) acc[mi][ni][j] = 0.0f;

  stage_tiles(A, B, M, N, K, bm, bn, 0, tid, ldsA[0], ldsB[0]);
  int pp = 0;
  for (int k0 = 0; k0 < K; k0 += BK) {
    __syncthreads();                           // buf[pp] ready; buf[pp^1] free
    if (k0 + BK < K) {
      __builtin_prefetch(A + (size_t)(bm + (tid >> 1)) * K + k0 + BK, 0, 1);
      stage_tiles(A, B, M, N, K, bm, bn, k0 + BK, tid, ldsA[pp ^ 1], ldsB[pp ^ 1]);
    }
    const unsigned short* la = ldsA[pp];
    const unsigned short* lb = ldsB[pp];

    FragBF bfr[2];
#pragma unroll
    for (int ni = 0; ni < 2; ++ni) {
      int nr = wn + ni * 16 + r;
      int sw = (nr & 3) << 3;
      bfr[ni].h[0] = *(const bf16x8*)(&lb[nr * BK + (kb ^ sw)]);
      bfr[ni].h[1] = *(const bf16x8*)(&lb[nr * BK + ((kb + 16) ^ sw)]);
    }
#pragma unroll
    for (int mi = 0; mi < 4; ++mi) {
      int ar = wm + mi * 16 + r;
      int sw = (ar & 3) << 3;
      FragBF af;
      af.h[0] = *(const bf16x8*)(&la[ar * BK + (kb ^ sw)]);
      af.h[1] = *(const bf16x8*)(&la[ar * BK + ((kb + 16) ^ sw)]);
#pragma unroll
      for (int ni = 0; ni < 2; ++ni)
        acc[mi][ni] = __builtin_amdgcn_wmma_f32_16x16x32_bf16(
            false, af.v, false, bfr[ni].v, (short)0, acc[mi][ni], false, false);
    }
    pp ^= 1;
  }

#pragma unroll
  for (int mi = 0; mi < 4; ++mi)
#pragma unroll
    for (int ni = 0; ni < 2; ++ni) {
      int col = bn + wn + ni * 16 + r;
#pragma unroll
      for (int j = 0; j < 8; ++j) {
        int row = bm + wm + mi * 16 + j + (hi << 3);
        if (row < M && col < N) {
          float v = acc[mi][ni][j];
          if (flags & 1) v += bias[col];
          if (flags & 2) v += res[(size_t)row * N + col];
          C[(size_t)row * N + col] = v;
        }
      }
    }
}

// ---------------- flash attention: one wave per (16 q-rows, head, batch) ----
// Q,K,V,O row-major [B*rows, DINNER], head offset h*DHEAD. Online softmax.
__global__ __launch_bounds__(32)
void k_flash(const float* __restrict__ Q, const float* __restrict__ K,
             const float* __restrict__ V, float* __restrict__ O, int T, int S)
{
  __shared__ unsigned short stg[16 * 32];   // P staging (one wave per block)
  const int lane = threadIdx.x & 31;
  const int r    = lane & 15;
  const int hi   = lane >> 4;
  const int kb   = hi << 3;
  const int hi16 = hi << 4;
  const int tb = blockIdx.x * 16;
  const int h  = blockIdx.y;
  const int b  = blockIdx.z;

  const float* qp = Q + ((size_t)(b * T + tb)) * DINNER + h * DHEAD;
  const float* kp = K + ((size_t)b * S) * DINNER + h * DHEAD;
  const float* vp = V + ((size_t)b * S) * DINNER + h * DHEAD;

  FragBF qf[2];                       // d=64 split into two K=32 slices
#pragma unroll
  for (int st = 0; st < 2; ++st)
#pragma unroll
    for (int i2 = 0; i2 < 4; ++i2) {
      const float* q0 = qp + (size_t)r * DINNER + st * 32;
      qf[st].w[i2]     = pack2bf(q0[kb + 2 * i2],      q0[kb + 2 * i2 + 1]);
      qf[st].w[4 + i2] = pack2bf(q0[kb + 16 + 2 * i2], q0[kb + 16 + 2 * i2 + 1]);
    }

  f32x8 o[4];
#pragma unroll
  for (int d = 0; d < 4; ++d)
#pragma unroll
    for (int j = 0; j < 8; ++j) o[d][j] = 0.f;
  float mrun[8], lrun[8];
#pragma unroll
  for (int j = 0; j < 8; ++j) { mrun[j] = -1e30f; lrun[j] = 0.f; }

  for (int s0 = 0; s0 < S; s0 += 32) {
    f32x8 sc[2];
#pragma unroll
    for (int c = 0; c < 2; ++c) {
#pragma unroll
      for (int j = 0; j < 8; ++j) sc[c][j] = 0.f;
      int s = s0 + c * 16 + r;
      bool ok = s < S;
#pragma unroll
      for (int st = 0; st < 2; ++st) {
        FragBF kf;
#pragma unroll
        for (int i2 = 0; i2 < 8; ++i2) {
          float x0 = 0.f, x1 = 0.f;
          if (ok) {
            const float* k0p = kp + (size_t)s * DINNER + st * 32 + kb + ((i2 >> 2) << 3);
            x0 = k0p[2 * i2]; x1 = k0p[2 * i2 + 1];
          }
          kf.w[i2] = pack2bf(x0, x1);
        }
        sc[c] = __builtin_amdgcn_wmma_f32_16x16x32_bf16(
            false, qf[st].v, false, kf.v, (short)0, sc[c], false, false);
      }
    }
    int sA = s0 + r, sB = s0 + 16 + r;
    float pa[8], pb[8];
#pragma unroll
    for (int j = 0; j < 8; ++j) {
      pa[j] = (sA < S) ? sc[0][j] * ATTNSCALE : -1e30f;
      pb[j] = (sB < S) ? sc[1][j] * ATTNSCALE : -1e30f;
    }
#pragma unroll
    for (int j = 0; j < 8; ++j) {
      float mj = redmax16(fmaxf(pa[j], pb[j]));
      float mn = fmaxf(mrun[j], mj);
      float scale = __expf(mrun[j] - mn);
      pa[j] = __expf(pa[j] - mn);
      pb[j] = __expf(pb[j] - mn);
      float ls = redsum16(pa[j] + pb[j]);
      lrun[j] = lrun[j] * scale + ls;
      mrun[j] = mn;
      o[0][j] *= scale; o[1][j] *= scale; o[2][j] *= scale; o[3][j] *= scale;
      stg[(j + (hi << 3)) * 32 + r]      = f2bf(pa[j]);
      stg[(j + (hi << 3)) * 32 + 16 + r] = f2bf(pb[j]);
    }
    __asm__ volatile("s_wait_dscnt 0x0" ::: "memory");
    FragBF pf;                                   // P as A-fragment (16x32)
    pf.h[0] = *(const bf16x8*)(&stg[r * 32 + kb]);
    pf.h[1] = *(const bf16x8*)(&stg[r * 32 + kb + 16]);
#pragma unroll
    for (int dd = 0; dd < 4; ++dd) {
      FragBF vf;
#pragma unroll
      for (int i2 = 0; i2 < 8; ++i2) {
        int s  = s0 + kb + 2 * i2 + ((i2 >> 2) << 3);
        float x0 = (s < S)     ? vp[(size_t)s * DINNER + dd * 16 + r]       : 0.f;
        float x1 = (s + 1 < S) ? vp[(size_t)(s + 1) * DINNER + dd * 16 + r] : 0.f;
        vf.w[i2] = pack2bf(x0, x1);
      }
      o[dd] = __builtin_amdgcn_wmma_f32_16x16x32_bf16(
          false, pf.v, false, vf.v, (short)0, o[dd], false, false);
    }
  }
#pragma unroll
  for (int j = 0; j < 8; ++j) {
    float inv = lrun[j] > 0.f ? 1.f / lrun[j] : 0.f;
    int row = b * T + tb + j + (hi << 3);
#pragma unroll
    for (int dd = 0; dd < 4; ++dd)
      O[(size_t)row * DINNER + h * DHEAD + dd * 16 + r] = o[dd][j] * inv;
  }
}

// ---------------- layernorm: one wave per row of HD ----------------
__global__ __launch_bounds__(256)
void k_ln(const float* __restrict__ x, const float* __restrict__ g,
          const float* __restrict__ bt, float* __restrict__ y, int rows)
{
  int wave = threadIdx.x >> 5, lane = threadIdx.x & 31;
  int row = blockIdx.x * 8 + wave;
  if (row >= rows) return;
  const float* xr = x + (size_t)row * HD;
  float v[40];
  float s = 0.f;
#pragma unroll
  for (int i = 0; i < 40; ++i) { v[i] = xr[lane + i * 32]; s += v[i]; }
  s = redsum32(s);
  float mean = s * (1.0f / HD);
  float q = 0.f;
#pragma unroll
  for (int i = 0; i < 40; ++i) { float d = v[i] - mean; q += d * d; }
  q = redsum32(q);
  float rstd = rsqrtf(q * (1.0f / HD) + LNEPS);
#pragma unroll
  for (int i = 0; i < 40; ++i) {
    int c = lane + i * 32;
    y[(size_t)row * HD + c] = (v[i] - mean) * rstd * g[c] + bt[c];
  }
}

// ---------------- input embedding + siren ----------------
__global__ __launch_bounds__(256)
void k_embed(const float* __restrict__ x, const float* __restrict__ coords,
             const float* __restrict__ ipw, const float* __restrict__ ipb,
             const float* __restrict__ sw,  const float* __restrict__ sb,
             float* __restrict__ cur)
{
  int row = blockIdx.x;                 // 0 .. B*N0-1
  float xv = x[row];
  float c0 = coords[(size_t)row * 3], c1 = coords[(size_t)row * 3 + 1],
        c2 = coords[(size_t)row * 3 + 2];
#pragma unroll
  for (int it = 0; it < 5; ++it) {
    int hh = it * 256 + threadIdx.x;
    float s = c0 * sw[hh] + c1 * sw[HD + hh] + c2 * sw[2 * HD + hh] + sb[hh];
    cur[(size_t)row * HD + hh] = xv * ipw[hh] + ipb[hh] + sinf(OMEGA0f * s);
  }
}

// ---------------- GEGLU ----------------
__global__ __launch_bounds__(256)
void k_geglu(const float* __restrict__ hdn, float* __restrict__ out, int rows)
{
  int row = blockIdx.x;
  int c = blockIdx.y * 256 + threadIdx.x;          // 0..5119
  float a = hdn[(size_t)row * (HD * 8) + c];
  float g = hdn[(size_t)row * (HD * 8) + HD * 4 + c];
  float ge = 0.5f * g * (1.0f + erff(g * 0.70710678f));
  out[(size_t)row * (HD * 4) + c] = a * ge;
}

// ---------------- fills / broadcast ----------------
__global__ void k_fillf(float* p, size_t n, float v) {
  size_t i = (size_t)blockIdx.x * 256 + threadIdx.x;
  if (i < n) p[i] = v;
}
__global__ void k_filli(int* p, int n, int v) {
  int i = blockIdx.x * 256 + threadIdx.x;
  if (i < n) p[i] = v;
}
__global__ void k_bcast(const float* __restrict__ src, float* __restrict__ dst) {
  int i = blockIdx.x * 256 + threadIdx.x;
  if (i < NBATCH * NLAT * HD) dst[i] = src[i % (NLAT * HD)];
}

// ---------------- downsample: coord min/max per batch ----------------
__global__ __launch_bounds__(256)
void k_minmax(const float* __restrict__ coords, float* __restrict__ mn,
              float* __restrict__ mx, int n)
{
  int b = blockIdx.x;
  const float* cp = coords + (size_t)b * n * 3;
  float lmn[3] = {1e30f, 1e30f, 1e30f}, lmx[3] = {-1e30f, -1e30f, -1e30f};
  for (int i = threadIdx.x; i < n; i += 256)
#pragma unroll
    for (int j = 0; j < 3; ++j) {
      float v = cp[(size_t)i * 3 + j];
      lmn[j] = fminf(lmn[j], v); lmx[j] = fmaxf(lmx[j], v);
    }
  __shared__ float smn[3][256], smx[3][256];
#pragma unroll
  for (int j = 0; j < 3; ++j) { smn[j][threadIdx.x] = lmn[j]; smx[j][threadIdx.x] = lmx[j]; }
  for (int st = 128; st; st >>= 1) {
    __syncthreads();
    if (threadIdx.x < st)
#pragma unroll
      for (int j = 0; j < 3; ++j) {
        smn[j][threadIdx.x] = fminf(smn[j][threadIdx.x], smn[j][threadIdx.x + st]);
        smx[j][threadIdx.x] = fmaxf(smx[j][threadIdx.x], smx[j][threadIdx.x + st]);
      }
  }
  __syncthreads();
  if (threadIdx.x == 0)
#pragma unroll
    for (int j = 0; j < 3; ++j) { mn[b * 3 + j] = smn[j][0]; mx[b * 3 + j] = smx[j][0]; }
}

// ---------------- downsample: scatter into cells ----------------
__global__ __launch_bounds__(256)
void k_assign(const float* __restrict__ cur, const float* __restrict__ coords,
              const float* __restrict__ mn, const float* __restrict__ mx,
              float* __restrict__ sums, float* __restrict__ counts,
              int* __restrict__ first, int n, int cpd, int c)
{
  int pt = blockIdx.x, b = blockIdx.y;
  const float* cc = coords + ((size_t)(b * n + pt)) * 3;
  int gi[3];
#pragma unroll
  for (int j = 0; j < 3; ++j) {
    float rng = mx[b * 3 + j] - mn[b * 3 + j] + 1e-6f;
    int g = (int)((cc[j] - mn[b * 3 + j]) / rng * (float)cpd);
    gi[j] = min(max(g, 0), cpd - 1);
  }
  int cell = gi[0] * cpd * cpd + gi[1] * cpd + gi[2];
  if (threadIdx.x == 0) {
    atomicAdd(&counts[b * c + cell], 1.0f);
    atomicMin(&first[b * c + cell], pt);
  }
  const float* xr = cur + ((size_t)(b * n + pt)) * HD;
  size_t base = ((size_t)b * c + cell) * HD;
#pragma unroll
  for (int it = 0; it < 5; ++it) {
    int ch = it * 256 + threadIdx.x;
    atomicAdd(&sums[base + ch], xr[ch]);
  }
}

// ---------------- stable occupied-cell compaction rank (exclusive scan) -----
__global__ __launch_bounds__(256)
void k_scan(const float* __restrict__ counts, int* __restrict__ rank,
            int* __restrict__ cnt, int c)
{
  int b = blockIdx.x;
  __shared__ int buf[256];
  __shared__ int carry;
  if (threadIdx.x == 0) carry = 0;
  __syncthreads();
  for (int base = 0; base < c; base += 256) {
    int i = base + threadIdx.x;
    int occ = (i < c && counts[b * c + i] > 0.0f) ? 1 : 0;
    buf[threadIdx.x] = occ;
    __syncthreads();
    for (int st = 1; st < 256; st <<= 1) {
      int v = (threadIdx.x >= st) ? buf[threadIdx.x - st] : 0;
      __syncthreads();
      buf[threadIdx.x] += v;
      __syncthreads();
    }
    if (i < c) rank[b * c + i] = carry + buf[threadIdx.x] - occ;
    __syncthreads();
    if (threadIdx.x == 255) carry += buf[255];
    __syncthreads();
  }
  if (threadIdx.x == 0) cnt[b] = carry;
}

// ---------------- gather compacted means + coords ----------------
__global__ __launch_bounds__(256)
void k_gather(const float* __restrict__ sums, const float* __restrict__ counts,
              const int* __restrict__ first, const int* __restrict__ rank,
              const float* __restrict__ coordsIn, float* __restrict__ means,
              float* __restrict__ coordsOut, int n, int c)
{
  int cell = blockIdx.x, b = blockIdx.y;
  float cv = counts[b * c + cell];
  if (cv <= 0.0f) return;
  int rk = rank[b * c + cell];
  float inv = 1.0f / cv;
#pragma unroll
  for (int it = 0; it < 5; ++it) {
    int ch = it * 256 + threadIdx.x;
    means[((size_t)b * c + rk) * HD + ch] =
        sums[((size_t)b * c + cell) * HD + ch] * inv;
  }
  if (threadIdx.x < 3) {
    int f = first[b * c + cell]; if (f > n - 1) f = n - 1;
    coordsOut[((size_t)b * c + rk) * 3 + threadIdx.x] =
        coordsIn[((size_t)(b * n + f)) * 3 + threadIdx.x];
  }
}

// zero rows >= occupied count (x_out * m in the reference)
__global__ __launch_bounds__(256)
void k_maskrows(float* __restrict__ x, const int* __restrict__ cnt, int c)
{
  int cell = blockIdx.x, b = blockIdx.y;
  if (cell < cnt[b]) return;
#pragma unroll
  for (int it = 0; it < 5; ++it)
    x[((size_t)b * c + cell) * HD + it * 256 + threadIdx.x] = 0.f;
}

// ================= host side =================
struct PLin { const float *w, *b; };
struct PLN  { const float *g, *b; };
struct PAtt { const float *wq, *wk, *wv; PLin out; };
struct PFF  { PLin l1, l2; };
struct PDec { PAtt attn; PFF ff; PLN fn, n_c, n_q; };
struct PEnc { PFF cff; PLN cffn, cn_c, cn_q; PAtt cross, sattn; PFF sff2; PLN sn1, sn2; };

static inline int cdiv(int a, int b) { return (a + b - 1) / b; }

static void gemm(hipStream_t s, const float* A, const float* B, const float* bias,
                 const float* res, float* C, int M, int N, int K, int flags) {
  dim3 g(cdiv(M, BM), cdiv(N, BN));
  k_gemm<<<g, dim3(256), 0, s>>>(A, B, bias, res, C, M, N, K, flags);
}

extern "C" void kernel_launch(void* const* d_in, const int* in_sizes, int n_in,
                              void* d_out, int out_size, void* d_ws, size_t ws_size,
                              hipStream_t stream) {
  (void)in_sizes; (void)n_in; (void)out_size; (void)ws_size;
  const float* x      = (const float*)d_in[0];
  const float* coords = (const float*)d_in[1];

  // ---- unpack params: jax pytree order (dicts sorted by key, lists in order)
  int ix = 2;
  auto nf = [&]() { return (const float*)d_in[ix++]; };
  PLin bbP;  bbP.b = nf();  bbP.w = nf();
  PLN  bbln; bbln.b = nf(); bbln.g = nf();
  PLin clP;  clP.b = nf();  clP.w = nf();
  PLN  clln; clln.b = nf(); clln.g = nf();
  PDec dec[3];
  for (int i = 0; i < 3; ++i) {
    PDec& d = dec[i];
    d.attn.out.b = nf(); d.attn.out.w = nf();
    d.attn.wk = nf(); d.attn.wq = nf(); d.attn.wv = nf();
    d.ff.l1.b = nf(); d.ff.l1.w = nf(); d.ff.l2.b = nf(); d.ff.l2.w = nf();
    d.fn.b = nf();  d.fn.g = nf();
    d.n_c.b = nf(); d.n_c.g = nf();
    d.n_q.b = nf(); d.n_q.g = nf();
  }
  PLin downP[3];
  for (int i = 0; i < 3; ++i) { downP[i].b = nf(); downP[i].w = nf(); }
  PEnc enc[4];
  for (int i = 0; i < 4; ++i) {
    PEnc& e = enc[i];
    e.cff.l1.b = nf(); e.cff.l1.w = nf(); e.cff.l2.b = nf(); e.cff.l2.w = nf();
    e.cffn.b = nf(); e.cffn.g = nf();
    e.cn_c.b = nf(); e.cn_c.g = nf();
    e.cn_q.b = nf(); e.cn_q.g = nf();
    e.cross.out.b = nf(); e.cross.out.w = nf();
    e.cross.wk = nf(); e.cross.wq = nf(); e.cross.wv = nf();
    e.sattn.out.b = nf(); e.sattn.out.w = nf();
    e.sattn.wk = nf(); e.sattn.wq = nf(); e.sattn.wv = nf();
    e.sff2.l1.b = nf(); e.sff2.l1.w = nf(); e.sff2.l2.b = nf(); e.sff2.l2.w = nf();
    e.sn1.b = nf(); e.sn1.g = nf();
    e.sn2.b = nf(); e.sn2.g = nf();
  }
  PLin ipP; ipP.b = nf(); ipP.w = nf();
  const float* latents = nf();
  PLin sirP; sirP.b = nf(); sirP.w = nf();
  PLin upP[3];
  for (int i = 0; i < 3; ++i) { upP[i].b = nf(); upP[i].w = nf(); }

  // ---- workspace arena ----
  char* wb = (char*)d_ws;
  size_t off = 0;
  auto af = [&](size_t n) { float* p = (float*)(wb + off);
                            off += ((n * 4 + 255) & ~(size_t)255); return p; };
  auto ai = [&](size_t n) { int* p = (int*)(wb + off);
                            off += ((n * 4 + 255) & ~(size_t)255); return p; };

  float* curL[4];
  for (int i = 0; i < 4; ++i) curL[i] = af((size_t)NBATCH * NLv[i] * HD);
  const float* crdL[4]; float* crdW[4];
  crdL[0] = coords;
  for (int i = 1; i < 4; ++i) { crdW[i] = af((size_t)NBATCH * NLv[i] * 3); crdL[i] = crdW[i]; }
  float* ctxn  = af((size_t)NBATCH * NLv[0] * HD);
  float* sums  = ctxn;                                    // alias (dead when used)
  float* means = ctxn + (size_t)NBATCH * 8000 * HD;       // alias
  float* kbuf = af((size_t)NBATCH * NLv[0] * DINNER);
  float* vbuf = af((size_t)NBATCH * NLv[0] * DINNER);
  float* qbuf = af((size_t)NBATCH * NLAT * DINNER);
  float* obuf = af((size_t)NBATCH * NLAT * DINNER);
  float* lat  = af((size_t)NBATCH * NLAT * HD);
  float* latn = af((size_t)NBATCH * NLAT * HD);
  float* skipUp  = af((size_t)NBATCH * NLAT * HD);
  float* skipUpN = af((size_t)NBATCH * NLAT * HD);
  float* skips[3];
  for (int i = 0; i < 3; ++i) skips[i] = af((size_t)NBATCH * NLAT * HD);
  float* hdn = af((size_t)NBATCH * NLAT * HD * 8);
  float* ggb = af((size_t)NBATCH * NLAT * HD * 4);
  float* mnb = af(NBATCH * 3);
  float* mxb = af(NBATCH * 3);
  float* cntf = af((size_t)NBATCH * 8000);
  int* firstI = ai((size_t)NBATCH * 8000);
  int* rankI  = ai((size_t)NBATCH * 8000);
  int* cntI   = ai(NBATCH);

  const int MROWS = NBATCH * NLAT;   // 512 latent rows

  auto LN = [&](const float* xin, const PLN& p, float* yout, int rows) {
    k_ln<<<cdiv(rows, 8), 256, 0, stream>>>(xin, p.g, p.b, yout, rows);
  };
  auto ATT = [&](const float* qn, const float* cn, int S, const PAtt& p) {
    gemm(stream, qn, p.wq, nullptr, nullptr, qbuf, MROWS, DINNER, HD, 0);
    gemm(stream, cn, p.wk, nullptr, nullptr, kbuf, NBATCH * S, DINNER, HD, 0);
    gemm(stream, cn, p.wv, nullptr, nullptr, vbuf, NBATCH * S, DINNER, HD, 0);
    k_flash<<<dim3(NLAT / 16, NHEADS, NBATCH), 32, 0, stream>>>(qbuf, kbuf, vbuf, obuf, NLAT, S);
    gemm(stream, obuf, p.out.w, p.out.b, lat, lat, MROWS, HD, DINNER, 3);
  };
  auto FFN = [&](const float* xn, const PFF& p) {
    gemm(stream, xn, p.l1.w, p.l1.b, nullptr, hdn, MROWS, HD * 8, HD, 1);
    k_geglu<<<dim3(MROWS, 20), 256, 0, stream>>>(hdn, ggb, MROWS);
    gemm(stream, ggb, p.l2.w, p.l2.b, lat, lat, MROWS, HD, HD * 4, 3);
  };
  auto FILLF = [&](float* p, size_t n, float v) {
    k_fillf<<<(unsigned)((n + 255) / 256), 256, 0, stream>>>(p, n, v);
  };
  auto DOWNS = [&](int lvl) {
    int n = NLv[lvl], cpd = CPDv[lvl], c = cpd * cpd * cpd;
    k_minmax<<<NBATCH, 256, 0, stream>>>(crdL[lvl], mnb, mxb, n);
    FILLF(sums, (size_t)NBATCH * c * HD, 0.f);
    FILLF(cntf, (size_t)NBATCH * c, 0.f);
    k_filli<<<cdiv(NBATCH * c, 256), 256, 0, stream>>>(firstI, NBATCH * c, n);
    k_assign<<<dim3(n, NBATCH), 256, 0, stream>>>(curL[lvl], crdL[lvl], mnb, mxb,
                                                  sums, cntf, firstI, n, cpd, c);
    k_scan<<<NBATCH, 256, 0, stream>>>(cntf, rankI, cntI, c);
    FILLF(means, (size_t)NBATCH * c * HD, 0.f);
    FILLF(crdW[lvl + 1], (size_t)NBATCH * c * 3, 0.f);
    k_gather<<<dim3(c, NBATCH), 256, 0, stream>>>(sums, cntf, firstI, rankI,
                                                  crdL[lvl], means, crdW[lvl + 1], n, c);
    gemm(stream, means, downP[lvl].w, downP[lvl].b, nullptr, curL[lvl + 1],
         NBATCH * c, HD, HD, 1);
    k_maskrows<<<dim3(c, NBATCH), 256, 0, stream>>>(curL[lvl + 1], cntI, c);
  };

  // ---- forward ----
  k_bcast<<<cdiv(NBATCH * NLAT * HD, 256), 256, 0, stream>>>(latents, lat);
  k_embed<<<NBATCH * NPTS0, 256, 0, stream>>>(x, coords, ipP.w, ipP.b,
                                              sirP.w, sirP.b, curL[0]);
  for (int i = 0; i < 4; ++i) {
    const PEnc& e = enc[i];
    LN(lat, e.cn_q, latn, MROWS);
    LN(curL[i], e.cn_c, ctxn, NBATCH * NLv[i]);
    ATT(latn, ctxn, NLv[i], e.cross);
    LN(lat, e.cffn, latn, MROWS);  FFN(latn, e.cff);
    LN(lat, e.sn1, latn, MROWS);   ATT(latn, latn, NLAT, e.sattn);
    LN(lat, e.sn2, latn, MROWS);   FFN(latn, e.sff2);
    if (i < 3) {
      hipMemcpyAsync(skips[i], lat, (size_t)MROWS * HD * 4,
                     hipMemcpyDeviceToDevice, stream);
      DOWNS(i);
    }
  }
  for (int i = 0; i < 3; ++i) {
    const PDec& d = dec[i];
    gemm(stream, skips[2 - i], upP[i].w, upP[i].b, nullptr, skipUp, MROWS, HD, HD, 1);
    LN(lat, d.n_q, latn, MROWS);
    LN(skipUp, d.n_c, skipUpN, MROWS);
    ATT(latn, skipUpN, NLAT, d.attn);
    LN(lat, d.fn, latn, MROWS);
    FFN(latn, d.ff);
  }
  float* outp = (float*)d_out;
  LN(lat, bbln, latn, MROWS);
  gemm(stream, latn, bbP.w, bbP.b, nullptr, outp, MROWS, ODIM, HD, 1);
  LN(lat, clln, latn, MROWS);
  gemm(stream, latn, clP.w, clP.b, nullptr, outp + (size_t)MROWS * ODIM,
       MROWS, ODIM, HD, 1);
}
